// ConvDatapath_32117765439784
// MI455X (gfx1250) — 8-bit matrix operand order repaired and checked by reading, compiled, not run on hardware
//
#include <hip/hip_runtime.h>
#include <stdint.h>

typedef __attribute__((ext_vector_type(8))) int   v8i;
typedef __attribute__((ext_vector_type(4))) int   v4i;
typedef __attribute__((ext_vector_type(8))) float v8f;

#define CFG_B 2
#define CIN   64
#define HH_   56
#define WW_   56
#define COUT  128
#define KDIM  576
#define NX    (CFG_B*HH_*WW_)
#define NB    5
#define NPB   116
#define KPAD  128
#define NSL   4

#define OFF_XP   ((size_t)0)
#define OFF_WP   ((size_t)16056320)
#define OFF_XS   ((size_t)16384000)
#define OFF_XO   ((size_t)16409088)
#define OFF_XSUM ((size_t)16434176)
#define OFF_WS   ((size_t)16459264)
#define OFF_WO   ((size_t)16459776)
#define OFF_WSUM ((size_t)16460288)

__global__ void prep_x_kernel(const float* __restrict__ x, uint8_t* __restrict__ xp,
                              float* __restrict__ xs, float* __restrict__ xo,
                              float* __restrict__ xsum) {
    const int wv   = threadIdx.x >> 5;
    const int lane = threadIdx.x & 31;
    const int row  = blockIdx.x * 8 + wv;
    const int b  = row / (HH_*WW_);
    const int hw = row - b*(HH_*WW_);
    const int h  = hw / WW_;
    const int w  = hw - h*WW_;

    float vals[18];
    float vmin = 1e30f, vmax = -1e30f, vsum = 0.f;
    #pragma unroll
    for (int t = 0; t < 18; ++t) {
        const int k  = lane + 32*t;
        const int ci = k / 9, r9 = k - ci*9;
        const int i  = r9 / 3, j = r9 - i*3;
        const int hr = h + i - 1, wr = w + j - 1;
        float v = 0.f;
        if (hr >= 0 && hr < HH_ && wr >= 0 && wr < WW_)
            v = x[((size_t)(b*CIN + ci)*HH_ + hr)*WW_ + wr];
        vals[t] = v;
        vmin = fminf(vmin, v); vmax = fmaxf(vmax, v); vsum += v;
    }
    for (int m = 16; m >= 1; m >>= 1) {
        vmin = fminf(vmin, __shfl_xor(vmin, m, 32));
        vmax = fmaxf(vmax, __shfl_xor(vmax, m, 32));
        vsum += __shfl_xor(vsum, m, 32);
    }
    const float step = (vmax - vmin) * (1.0f/255.0f);
    const float inv  = step > 0.f ? 1.0f/step : 0.f;
    if (lane == 0) { xs[row] = step; xo[row] = vmin; xsum[row] = vsum; }

    #pragma unroll
    for (int t = 0; t < 18; ++t) {
        const int k = lane + 32*t;
        int q = (int)rintf((vals[t] - vmin) * inv);
        q = q < 0 ? 0 : (q > 255 ? 255 : q);
        const int c = k / NPB, pos = k - c*NPB;
        #pragma unroll
        for (int s = 0; s < NSL; ++s)
            xp[(((size_t)(s*NB + c)*NX) + row)*KPAD + pos] = (uint8_t)((q >> (6 - 2*s)) & 3);
    }
    for (int idx = lane; idx < NSL*NB*16; idx += 32) {
        const int s = idx / (NB*16), rem = idx - s*NB*16;
        const int c = rem >> 4, p = 112 + (rem & 15);
        const int cov = (c < NB-1) ? NPB : (KDIM - (NB-1)*NPB);
        if (p >= cov)
            xp[(((size_t)(s*NB + c)*NX) + row)*KPAD + p] = 0;
    }
}

__global__ void prep_w_kernel(const float* __restrict__ wgt, uint8_t* __restrict__ wp,
                              float* __restrict__ wsc, float* __restrict__ wo,
                              float* __restrict__ wsum) {
    const int wv   = threadIdx.x >> 5;
    const int lane = threadIdx.x & 31;
    const int row  = blockIdx.x * 4 + wv;
    float vals[18];
    float vmin = 1e30f, vmax = -1e30f, vsum = 0.f;
    #pragma unroll
    for (int t = 0; t < 18; ++t) {
        const int k = lane + 32*t;
        const float v = wgt[(size_t)row*KDIM + k];
        vals[t] = v;
        vmin = fminf(vmin, v); vmax = fmaxf(vmax, v); vsum += v;
    }
    for (int m = 16; m >= 1; m >>= 1) {
        vmin = fminf(vmin, __shfl_xor(vmin, m, 32));
        vmax = fmaxf(vmax, __shfl_xor(vmax, m, 32));
        vsum += __shfl_xor(vsum, m, 32);
    }
    const float step = (vmax - vmin) * (1.0f/255.0f);
    const float inv  = step > 0.f ? 1.0f/step : 0.f;
    if (lane == 0) { wsc[row] = step; wo[row] = vmin; wsum[row] = vsum; }

    #pragma unroll
    for (int t = 0; t < 18; ++t) {
        const int k = lane + 32*t;
        int q = (int)rintf((vals[t] - vmin) * inv);
        q = q < 0 ? 0 : (q > 255 ? 255 : q);
        const int c = k / NPB, pos = k - c*NPB;
        #pragma unroll
        for (int s = 0; s < NSL; ++s)
            wp[(((size_t)(s*NB + c)*COUT) + row)*KPAD + pos] = (uint8_t)((q >> (6 - 2*s)) & 3);
    }
    for (int idx = lane; idx < NSL*NB*16; idx += 32) {
        const int s = idx / (NB*16), rem = idx - s*NB*16;
        const int c = rem >> 4, p = 112 + (rem & 15);
        const int cov = (c < NB-1) ? NPB : (KDIM - (NB-1)*NPB);
        if (p >= cov)
            wp[(((size_t)(s*NB + c)*COUT) + row)*KPAD + p] = 0;
    }
}

__global__ __launch_bounds__(256, 1)
void crossbar_gemm_kernel(const uint8_t* __restrict__ xp, const uint8_t* __restrict__ wp,
                          const float* __restrict__ xs, const float* __restrict__ xo,
                          const float* __restrict__ xsum,
                          const float* __restrict__ wsc, const float* __restrict__ wo,
                          const float* __restrict__ wsum,
                          float* __restrict__ out) {
    const int wv    = threadIdx.x >> 5;
    const int lane  = threadIdx.x & 31;
    const int lo    = lane & 15;
    const int hi    = lane >> 4;
    const int xtile = blockIdx.x;
    const int xrowA = xtile*16 + lo;
    const int y     = wv*16 + lo;

    v8f facc;
    #pragma unroll
    for (int e = 0; e < 8; ++e) facc[e] = 0.f;

    const float sliceW[4] = {64.f, 16.f, 4.f, 1.f};

    for (int c = 0; c < NB; ++c) {
        v8i A[NSL][2], Bm[NSL][2];
        #pragma unroll
        for (int s = 0; s < NSL; ++s) {
            const uint8_t* ab = xp + (((size_t)(s*NB + c)*NX)   + xrowA)*KPAD;
            const uint8_t* bb = wp + (((size_t)(s*NB + c)*COUT) + y    )*KPAD;
            #pragma unroll
            for (int hf = 0; hf < 2; ++hf) {
                const int kb = hf*64;
                union { v8i v; unsigned long long u[4]; } af;
                #pragma unroll
                for (int j = 0; j < 4; ++j)
                    af.u[j] = *(const unsigned long long*)(ab + kb + 32*(j>>1) + 16*hi + 8*(j&1));
                A[s][hf] = af.v;
                union { v8i v; v4i q[2]; } bf;
                #pragma unroll
                for (int g = 0; g < 2; ++g)
                    bf.q[g] = *(const v4i*)(bb + kb + 32*g + 16*hi);
                Bm[s][hf] = bf.v;
            }
        }
        #pragma unroll
        for (int ws = 0; ws < NSL; ++ws) {
            #pragma unroll
            for (int is = 0; is < NSL; ++is) {
                v8i acc = {};
                acc = __builtin_amdgcn_wmma_i32_16x16x64_iu8(
                        false, A[is][0], false, Bm[ws][0], acc, false, false);
                acc = __builtin_amdgcn_wmma_i32_16x16x64_iu8(
                        false, A[is][1], false, Bm[ws][1], acc, false, false);
                const float wfac = sliceW[ws] * sliceW[is];
                #pragma unroll
                for (int e = 0; e < 8; ++e) {
                    int z = acc[e];
                    z = z > 1024 ? 1024 : (z < 0 ? 0 : z);
                    facc[e] += wfac * (rintf((float)z * 0.25f) * 4.0f);
                }
            }
        }
    }

    const float sy = wsc[y], oy = wo[y], smy = wsum[y];
    #pragma unroll
    for (int e = 0; e < 8; ++e) {
        const int xr = xtile*16 + e + 8*hi;
        const float sx = xs[xr], ox = xo[xr], smx = xsum[xr];
        const float val = facc[e]*sx*sy + ox*smy + oy*smx - ox*oy*(float)KDIM;
        const int b  = xr / (HH_*WW_);
        const int hw = xr - b*(HH_*WW_);
        out[((size_t)b*COUT + y)*(HH_*WW_) + hw] = val;
    }
}

extern "C" void kernel_launch(void* const* d_in, const int* in_sizes, int n_in,
                              void* d_out, int out_size, void* d_ws, size_t ws_size,
                              hipStream_t stream) {
    const float* x = (const float*)d_in[0];
    const float* w = (const float*)d_in[1];
    float* out = (float*)d_out;
    uint8_t* ws = (uint8_t*)d_ws;

    uint8_t* xpk  = ws + OFF_XP;
    uint8_t* wpk  = ws + OFF_WP;
    float* xs   = (float*)(ws + OFF_XS);
    float* xo   = (float*)(ws + OFF_XO);
    float* xsm  = (float*)(ws + OFF_XSUM);
    float* wsc  = (float*)(ws + OFF_WS);
    float* wo   = (float*)(ws + OFF_WO);
    float* wsm  = (float*)(ws + OFF_WSUM);

    prep_x_kernel<<<NX/8, 256, 0, stream>>>(x, xpk, xs, xo, xsm);
    prep_w_kernel<<<COUT/4, 128, 0, stream>>>(w, wpk, wsc, wo, wsm);
    crossbar_gemm_kernel<<<NX/16, 256, 0, stream>>>(xpk, wpk, xs, xo, xsm,
                                                    wsc, wo, wsm, out);
}
